// DiscrepancyVAE_90975997264267
// MI455X (gfx1250) — compile-verified, matrix operand order checked by reading, not run on hardware
//
#include <hip/hip_runtime.h>
#include <hip/hip_bf16.h>

typedef __attribute__((ext_vector_type(16))) _Float16 v16h;
typedef __attribute__((ext_vector_type(8)))  float    v8f;
typedef unsigned int u32x4 __attribute__((ext_vector_type(4)));
typedef int          i32x8 __attribute__((ext_vector_type(8)));
typedef int          i32x4 __attribute__((ext_vector_type(4)));

#define GENES 2000
#define CELLS 64
#define NNODE (GENES * CELLS)      // 128000
#define NEDGE (NNODE * 8)          // 1024000
#define H1    256
#define H2    128
#define LAT   32
#define DH1   128
#define DH2   256
#define OUTG  2000

#define KPITCH 264                 // padded K pitch for W2t (f16): 132 dwords -> bank stride 4
#define APITCH 260                 // padded K pitch for d2 (f32): bank stride 4
#define MBLK   8                   // M-tiles per block in k_gemm1

// ---- workspace layout (float offsets) ----
#define WS_DEGF   0
#define WS_DINV   (WS_DEGF   + NNODE)
#define WS_INVDEG (WS_DINV   + NNODE)
#define WS_S      (WS_INVDEG + NNODE)
#define WS_T      (WS_S      + NNODE)
#define WS_AGG2   (WS_T      + NNODE)
#define WS_POOL   (WS_AGG2   + NNODE * H2)     // 17,024,000
#define WS_Z      (WS_POOL   + CELLS * H2)
#define WS_D1     (WS_Z      + CELLS * LAT)
#define WS_D2     (WS_D1     + CELLS * DH1)    // padded [64][APITCH] f32
#define WS_W2T    (WS_D2     + CELLS * APITCH) // [128][KPITCH] f16 = 16896 floats

// ---- d_out layout (float offsets) ----
#define OUT_RECON 0
#define OUT_MU    (OUT_RECON + CELLS * OUTG)   // 128000
#define OUT_LV    (OUT_MU    + CELLS * LAT)    // 130048
#define OUT_GE    (OUT_LV    + CELLS * LAT)    // 132096

// K index for element e (0..15) of a 16-bit WMMA A/B operand, given lane-half (ISA 7.12.2).
__device__ __forceinline__ int wmma_k(int e, int half) {
    int v = e >> 1;
    return ((v < 4) ? 0 : 16) + half * 8 + 2 * (v & 3) + (e & 1);
}

// 1-D TDM copy: nelem elements of (1<<log2sz) bytes, global -> LDS byte offset lds_off.
// D# per ISA 8.3/8.4: group0 = {ctrl(count=1), lds_addr, gaddr_lo, gaddr_hi|type=2},
// group1 packs data_size / tensor_dim0 / tensor_dim1=1 / tile_dim0 / stride0.
__device__ __forceinline__ void tdm_load_1d(unsigned lds_off, const void* gptr,
                                            unsigned nelem, int log2sz) {
    unsigned long long ga = (unsigned long long)(size_t)gptr;
    u32x4 g0;
    g0.x = 1u;                                   // count=1, user mode
    g0.y = lds_off;                              // LDS byte address
    g0.z = (unsigned)ga;                         // global_addr[31:0]
    g0.w = (unsigned)(ga >> 32) | (2u << 30);    // global_addr[56:32] | type=2
    i32x8 g1;
    g1[0] = (int)((unsigned)log2sz << 16);       // data_size; wg_mask=0 (not in cluster)
    g1[1] = (int)((nelem & 0xFFFFu) << 16);      // tensor_dim0[15:0] (bits 63:48)
    g1[2] = (int)(((nelem >> 16) & 0xFFFFu) | (1u << 16)); // dim0[31:16], tensor_dim1=1
    g1[3] = (int)((nelem & 0xFFFFu) << 16);      // tile_dim0 = nelem (bits 127:112)
    g1[4] = 0;                                   // tile_dim1=0 (1-D tile), tile_dim2=0
    g1[5] = (int)nelem;                          // tensor_dim0_stride[31:0]
    g1[6] = 0;
    g1[7] = 0;
    i32x4 zz = {0, 0, 0, 0};
#if __clang_major__ >= 23
    i32x8 z8 = {0, 0, 0, 0, 0, 0, 0, 0};
    __builtin_amdgcn_tensor_load_to_lds(g0, g1, zz, zz, z8, 0);
#else
    __builtin_amdgcn_tensor_load_to_lds(g0, g1, zz, zz, 0);
#endif
}

// ---------------- setup kernels ----------------
__global__ void k_zero(float* ws, int n) {
    int i = blockIdx.x * blockDim.x + threadIdx.x;
    if (i < n) ws[i] = 0.0f;
}

__global__ void k_deg(const int* __restrict__ dst, float* __restrict__ degf) {
    int e = blockIdx.x * blockDim.x + threadIdx.x;
    if (e < NEDGE) unsafeAtomicAdd(&degf[dst[e]], 1.0f);
}

__global__ void k_dinv(const float* __restrict__ degf,
                       float* __restrict__ dinv, float* __restrict__ invdeg) {
    int i = blockIdx.x * blockDim.x + threadIdx.x;
    if (i < NNODE) {
        float d = degf[i] + 1.0f;            // self loop
        dinv[i]   = rsqrtf(d);
        invdeg[i] = 1.0f / d;
    }
}

// layer-1 aggregation is rank-1: only a scalar per node is scattered
__global__ void k_s(const int* __restrict__ src, const int* __restrict__ dst,
                    const float* __restrict__ x, const float* __restrict__ dinv,
                    float* __restrict__ s) {
    int e = blockIdx.x * blockDim.x + threadIdx.x;
    if (e < NEDGE) {
        int si = src[e];
        unsafeAtomicAdd(&s[dst[e]], x[si] * dinv[si]);
    }
}

__global__ void k_t(const float* __restrict__ x, const float* __restrict__ s,
                    const float* __restrict__ dinv, const float* __restrict__ invdeg,
                    float* __restrict__ t) {
    int i = blockIdx.x * blockDim.x + threadIdx.x;
    if (i < NNODE) t[i] = dinv[i] * s[i] + x[i] * invdeg[i];
}

// transpose+convert W2 [256x128] f32 -> W2t [128][KPITCH] f16 (pad uninitialized, never read)
__global__ void k_cvtW2(const float* __restrict__ W2, _Float16* __restrict__ W2t) {
    int gid = blockIdx.x * blockDim.x + threadIdx.x;   // H2*H1 = 32768
    if (gid >= H2 * H1) return;
    const int k = gid & (H1 - 1);
    const int n = gid >> 8;
    W2t[n * KPITCH + k] = (_Float16)W2[k * H2 + n];
}

// ------- big GEMM: xw2 = relu(t*W1+b1) @ W2  [128000x256 @ 256x128] -------
// Block = 8 waves; wave wv owns N-tile wv; block iterates MBLK M-tiles, reusing
// the TDM-staged LDS copy of W2t. A is regenerated on the fly from scalar t.
__global__ __launch_bounds__(256) void k_gemm1(
        const float* __restrict__ t,  const float* __restrict__ W1,
        const float* __restrict__ b1, const _Float16* __restrict__ W2t,
        float* __restrict__ xw2) {
    __shared__ _Float16 w2s[H2 * KPITCH];   // 67.6 KB
    __shared__ float w1s[H1], b1s[H1];
    const int tid = threadIdx.x;
    for (int i = tid; i < H1; i += 256) { w1s[i] = W1[i]; b1s[i] = b1[i]; }
    if (tid < 32) {                         // wave 0 drives the TDM
        tdm_load_1d((unsigned)(size_t)w2s, W2t, H2 * KPITCH, 1);
        __builtin_amdgcn_s_wait_tensorcnt(0);
    }
    __syncthreads();

    const int lane = tid & 31;
    const int wv   = tid >> 5;              // n-tile 0..7
    const int mrow = lane & 15;
    const int half = lane >> 4;
    const int ncol = wv * 16 + mrow;
    const _Float16* bcol = &w2s[ncol * KPITCH];

    for (int mb = 0; mb < MBLK; ++mb) {
        const int m0 = (blockIdx.x * MBLK + mb) * 16;
        const float tm = t[m0 + mrow];
        v8f acc = {};
#pragma unroll
        for (int ks = 0; ks < 8; ++ks) {    // K = 256
            const int kb = ks * 32;
            v16h a, b;
#pragma unroll
            for (int e = 0; e < 16; ++e) {
                const int k = kb + wmma_k(e, half);
                a[e] = (_Float16)fmaxf(fmaf(tm, w1s[k], b1s[k]), 0.0f);
                b[e] = bcol[k];
            }
            acc = __builtin_amdgcn_wmma_f32_16x16x32_f16(
                    false, a, false, b, (short)0, acc, false, false);
        }
#pragma unroll
        for (int r = 0; r < 8; ++r)
            xw2[(size_t)(m0 + r + 8 * half) * H2 + ncol] = acc[r];
    }
}

// ------- layer-2 edge scatter: agg2[dst] += xw2[src] * coef -------
__global__ void k_scatter(const int* __restrict__ src, const int* __restrict__ dst,
                          const float* __restrict__ dinv,
                          const float* __restrict__ xw2, float* __restrict__ agg2) {
    int gid = blockIdx.x * blockDim.x + threadIdx.x;   // NEDGE * 32 threads
    if (gid >= NEDGE * 32) return;
    const int e  = gid >> 5;
    const int fg = (gid & 31) * 4;
    const int si = src[e], di = dst[e];
    const float c = dinv[si] * dinv[di];
    const float4 v = *(const float4*)&xw2[(size_t)si * H2 + fg];
    float* p = &agg2[(size_t)di * H2 + fg];
    unsafeAtomicAdd(p + 0, v.x * c);
    unsafeAtomicAdd(p + 1, v.y * c);
    unsafeAtomicAdd(p + 2, v.z * c);
    unsafeAtomicAdd(p + 3, v.w * c);
}

// gene_emb = agg2 + xw2/deg + b2  (xw2 staged in-place in the gene_emb output region)
__global__ void k_combine(const float* __restrict__ agg2, const float* __restrict__ invdeg,
                          const float* __restrict__ b2, float* __restrict__ ge) {
    int gid = blockIdx.x * blockDim.x + threadIdx.x;   // NNODE*H2
    if (gid >= NNODE * H2) return;
    const int i = gid >> 7;
    const int f = gid & 127;
    ge[gid] = agg2[gid] + ge[gid] * invdeg[i] + b2[f];
}

// per-cell mean pool of relu(gene_emb)
__global__ void k_pool(const float* __restrict__ ge, float* __restrict__ pooled) {
    const int c = blockIdx.x, f = threadIdx.x;         // 64 x 128
    const float* base = ge + (size_t)c * GENES * H2 + f;
    float sum = 0.0f;
    for (int g = 0; g < GENES; ++g) sum += fmaxf(base[(size_t)g * H2], 0.0f);
    pooled[c * H2 + f] = sum * (1.0f / (float)GENES);
}

// mu / logvar / reparameterized z
__global__ void k_head(const float* __restrict__ pooled,
                       const float* __restrict__ Wmu, const float* __restrict__ bmu,
                       const float* __restrict__ Wlv, const float* __restrict__ blv,
                       const float* __restrict__ eps,
                       float* __restrict__ mu, float* __restrict__ lv, float* __restrict__ z) {
    int gid = blockIdx.x * blockDim.x + threadIdx.x;   // CELLS*LAT = 2048
    if (gid >= CELLS * LAT) return;
    const int c = gid >> 5, l = gid & 31;
    float pm = bmu[l], pv = blv[l];
    for (int k = 0; k < H2; ++k) {
        const float p = pooled[c * H2 + k];
        pm = fmaf(p, Wmu[k * LAT + l], pm);
        pv = fmaf(p, Wlv[k * LAT + l], pv);
    }
    mu[gid] = pm;
    lv[gid] = pv;
    z[gid]  = pm + eps[gid] * expf(0.5f * pv);
}

__global__ void k_dec1(const float* __restrict__ z, const float* __restrict__ Wd1,
                       const float* __restrict__ bd1, float* __restrict__ d1) {
    int gid = blockIdx.x * blockDim.x + threadIdx.x;   // CELLS*DH1 = 8192
    if (gid >= CELLS * DH1) return;
    const int c = gid >> 7, f = gid & 127;
    float acc = bd1[f];
    for (int k = 0; k < LAT; ++k) acc = fmaf(z[c * LAT + k], Wd1[k * DH1 + f], acc);
    d1[gid] = fmaxf(acc, 0.0f);
}

// writes d2 with padded pitch APITCH so it can be TDM-staged conflict-free
__global__ void k_dec2(const float* __restrict__ d1, const float* __restrict__ Wd2,
                       const float* __restrict__ bd2, float* __restrict__ d2p) {
    int gid = blockIdx.x * blockDim.x + threadIdx.x;   // CELLS*DH2 = 16384
    if (gid >= CELLS * DH2) return;
    const int c = gid >> 8, f = gid & 255;
    float acc = bd2[f];
    for (int k = 0; k < DH1; ++k) acc = fmaf(d1[c * DH1 + k], Wd2[k * DH2 + f], acc);
    d2p[c * APITCH + f] = fmaxf(acc, 0.0f);
}

// ------- recon = d2 @ Wout + bout  [64x256 @ 256x2000] via WMMA -------
// d2 (66.5 KB padded) TDM-staged in LDS; 500 tiles, 8 waves/block, uniform early-out.
__global__ __launch_bounds__(256) void k_gemm_out(
        const float* __restrict__ d2p, const float* __restrict__ Wout,
        const float* __restrict__ bout, float* __restrict__ recon) {
    __shared__ float d2s[CELLS * APITCH];
    if (threadIdx.x < 32) {
        tdm_load_1d((unsigned)(size_t)d2s, d2p, CELLS * APITCH, 2);
        __builtin_amdgcn_s_wait_tensorcnt(0);
    }
    __syncthreads();

    const int lane = threadIdx.x & 31;
    const int tile = blockIdx.x * 8 + (threadIdx.x >> 5);
    if (tile >= 4 * 125) return;                  // uniform per wave
    const int mt = tile / 125, nt = tile % 125;
    const int mrow = lane & 15;
    const int half = lane >> 4;
    const int m0   = mt * 16;
    const int ncol = nt * 16 + mrow;
    const float* arow = &d2s[(m0 + mrow) * APITCH];
    v8f acc = {};
#pragma unroll
    for (int ks = 0; ks < 8; ++ks) {              // K = 256
        const int kb = ks * 32;
        v16h a, b;
#pragma unroll
        for (int e = 0; e < 16; ++e) {
            const int k = kb + wmma_k(e, half);
            a[e] = (_Float16)arow[k];
            b[e] = (_Float16)Wout[k * OUTG + ncol];
        }
        acc = __builtin_amdgcn_wmma_f32_16x16x32_f16(
                false, a, false, b, (short)0, acc, false, false);
    }
    const float bb = bout[ncol];
#pragma unroll
    for (int r = 0; r < 8; ++r)
        recon[(size_t)(m0 + r + 8 * half) * OUTG + ncol] = acc[r] + bb;
}

// ---------------- launcher ----------------
extern "C" void kernel_launch(void* const* d_in, const int* in_sizes, int n_in,
                              void* d_out, int out_size, void* d_ws, size_t ws_size,
                              hipStream_t stream) {
    const float* x    = (const float*)d_in[0];
    const int*   ei   = (const int*)  d_in[1];   // [2, NEDGE]
    const float* eps  = (const float*)d_in[3];
    const float* W1   = (const float*)d_in[4];
    const float* b1   = (const float*)d_in[5];
    const float* W2   = (const float*)d_in[6];
    const float* b2   = (const float*)d_in[7];
    const float* Wmu  = (const float*)d_in[8];
    const float* bmu  = (const float*)d_in[9];
    const float* Wlv  = (const float*)d_in[10];
    const float* blv  = (const float*)d_in[11];
    const float* Wd1  = (const float*)d_in[12];
    const float* bd1  = (const float*)d_in[13];
    const float* Wd2  = (const float*)d_in[14];
    const float* bd2  = (const float*)d_in[15];
    const float* Wout = (const float*)d_in[16];
    const float* bout = (const float*)d_in[17];

    const int* src = ei;
    const int* dst = ei + NEDGE;

    float* ws  = (float*)d_ws;
    float* out = (float*)d_out;
    float* recon = out + OUT_RECON;
    float* mu    = out + OUT_MU;
    float* lv    = out + OUT_LV;
    float* ge    = out + OUT_GE;      // xw2 staged here, combined in place

    float*     degf   = ws + WS_DEGF;
    float*     dinv   = ws + WS_DINV;
    float*     invdeg = ws + WS_INVDEG;
    float*     s      = ws + WS_S;
    float*     t      = ws + WS_T;
    float*     agg2   = ws + WS_AGG2;
    float*     pooled = ws + WS_POOL;
    float*     z      = ws + WS_Z;
    float*     d1     = ws + WS_D1;
    float*     d2p    = ws + WS_D2;
    _Float16*  W2t    = (_Float16*)(ws + WS_W2T);

    const int B = 256;
    // zero accumulators (degf, s, agg2)
    k_zero<<<(WS_POOL + B - 1) / B, B, 0, stream>>>(ws, WS_POOL);
    // degrees + normalization
    k_deg  <<<(NEDGE + B - 1) / B, B, 0, stream>>>(dst, degf);
    k_dinv <<<(NNODE + B - 1) / B, B, 0, stream>>>(degf, dinv, invdeg);
    // layer-1 rank-1 collapse
    k_s    <<<(NEDGE + B - 1) / B, B, 0, stream>>>(src, dst, x, dinv, s);
    k_t    <<<(NNODE + B - 1) / B, B, 0, stream>>>(x, s, dinv, invdeg, t);
    // W2 -> f16 transposed/padded, then WMMA GEMM with TDM-staged LDS weights
    k_cvtW2<<<(H2 * H1 + B - 1) / B, B, 0, stream>>>(W2, W2t);
    k_gemm1<<<NNODE / (16 * MBLK), B, 0, stream>>>(t, W1, b1, W2t, ge);
    // layer-2 aggregation + combine
    k_scatter<<<(NEDGE * 32 + B - 1) / B, B, 0, stream>>>(src, dst, dinv, ge, agg2);
    k_combine<<<(NNODE * H2 + B - 1) / B, B, 0, stream>>>(agg2, invdeg, b2, ge);
    // pool + VAE head + decoder
    k_pool<<<CELLS, H2, 0, stream>>>(ge, pooled);
    k_head<<<(CELLS * LAT + B - 1) / B, B, 0, stream>>>(pooled, Wmu, bmu, Wlv, blv, eps,
                                                       mu, lv, z);
    k_dec1<<<(CELLS * DH1 + B - 1) / B, B, 0, stream>>>(z, Wd1, bd1, d1);
    k_dec2<<<(CELLS * DH2 + B - 1) / B, B, 0, stream>>>(d1, Wd2, bd2, d2p);
    // final decoder GEMM (WMMA + TDM-staged activations)
    k_gemm_out<<<(500 + 7) / 8, B, 0, stream>>>(d2p, Wout, bout, recon);
}
